// MixtralMoE_26379689132539
// MI455X (gfx1250) — matrix operand order repaired and checked by reading, not compiled, not run on hardware
//
#include <hip/hip_runtime.h>
#include <hip/hip_bf16.h>

// Problem constants (reference: B=1, S=4096, H=1024, I=2048, E=8, TOP_K=2)
constexpr int kT = 4096;   // tokens
constexpr int kH = 1024;   // hidden
constexpr int kI = 2048;   // intermediate
constexpr int kE = 8;      // experts
constexpr int kM = 32;     // token tile (two 16-row WMMA halves)
constexpr int kPad = 8;    // LDS row pad (halves): row stride ≡ 4 dwords mod 64 banks
constexpr int kLdx = kH + kPad;   // X tile LDS row stride (halves)
constexpr int kLda = kI + kPad;   // act tile LDS row stride (halves)

typedef __attribute__((ext_vector_type(16))) __bf16        v16bf;
typedef __attribute__((ext_vector_type(8)))  float         v8f;
typedef __attribute__((ext_vector_type(8)))  unsigned int  v8u;

// ---------- helpers ----------

__device__ __forceinline__ unsigned int pack2bf(float a, float b) {
#if __has_builtin(__builtin_amdgcn_cvt_pk_bf16_f32)
  typedef __attribute__((ext_vector_type(2))) __bf16 v2bf;
  v2bf r = __builtin_amdgcn_cvt_pk_bf16_f32(a, b);
  return __builtin_bit_cast(unsigned int, r);
#else
  union { float f; unsigned int u; } va, vb;
  va.f = a; vb.f = b;
  unsigned int ua = va.u + (0x7FFFu + ((va.u >> 16) & 1u));
  unsigned int ub = vb.u + (0x7FFFu + ((vb.u >> 16) & 1u));
  return (ua >> 16) | (ub & 0xFFFF0000u);
#endif
}

__device__ __forceinline__ unsigned short f2bf(float f) {
  return (unsigned short)(pack2bf(f, 0.0f) & 0xFFFFu);
}

__device__ __forceinline__ float fast_rcp(float x) {
#if __has_builtin(__builtin_amdgcn_rcpf)
  return __builtin_amdgcn_rcpf(x);   // v_rcp_f32 (approx, fine before bf16 truncation)
#else
  return 1.0f / x;
#endif
}

__device__ __forceinline__ v8f wmma_bf16(v8u a, v8u b, v8f c) {
  return __builtin_amdgcn_wmma_f32_16x16x32_bf16(
      /*neg_a=*/false, __builtin_bit_cast(v16bf, a),
      /*neg_b=*/false, __builtin_bit_cast(v16bf, b),
      /*c_mod=*/(short)0, c, /*reuse_a=*/false, /*reuse_b=*/false);
}

// A-matrix fragment (16x32 bf16) from a row-major (padded) LDS tile.
// ISA 7.12.2: lanes 0-15 hold row M=lane, K 0-7 (v0-3) and 16-23 (v4-7);
// lanes 16-31 hold row M=lane-16, K 8-15 and 24-31.
__device__ __forceinline__ v8u load_a_lds(const unsigned short* base, int ldh,
                                          int row, int kb, int lane) {
  const int koff = (lane < 16) ? 0 : 8;
  const unsigned short* p = base + row * ldh + kb + koff;
  uint4 lo = *(const uint4*)p;          // K kb+koff    .. +7
  uint4 hi = *(const uint4*)(p + 16);   // K kb+koff+16 .. +23
  v8u r;
  r[0] = lo.x; r[1] = lo.y; r[2] = lo.z; r[3] = lo.w;
  r[4] = hi.x; r[5] = hi.y; r[6] = hi.z; r[7] = hi.w;
  return r;
}

// B-matrix fragment (32x16 bf16) from pre-converted bf16 weights, K-major rows.
// lanes 0-15 hold col N=lane, K 0-7 (v0-3) and 16-23 (v4-7);
// lanes 16-31 hold col N=lane-16, K 8-15 and 24-31.  wrow = &W[(n0+lane%16) * ld].
__device__ __forceinline__ v8u load_b_bf(const unsigned short* wrow, int kb, int lane) {
  const unsigned short* p = wrow + kb + ((lane < 16) ? 0 : 8);
  uint4 lo = *(const uint4*)p;        // K +0..7
  uint4 hi = *(const uint4*)(p + 16); // K +16..23
  v8u r;
  r[0] = lo.x; r[1] = lo.y; r[2] = lo.z; r[3] = lo.w;
  r[4] = hi.x; r[5] = hi.y; r[6] = hi.z; r[7] = hi.w;
  return r;
}

// ---------- kernel 0: fp32 -> bf16 bulk conversion (one pass) ----------

__global__ __launch_bounds__(256)
void moe_cvt_kernel(const float* __restrict__ in, unsigned short* __restrict__ out,
                    int n4 /* count of float4 */) {
  const int i = blockIdx.x * 256 + threadIdx.x;
  if (i >= n4) return;
  const float4 v = ((const float4*)in)[i];
  uint2 p;
  p.x = pack2bf(v.x, v.y);
  p.y = pack2bf(v.z, v.w);
  ((uint2*)out)[i] = p;
}

// ---------- kernel 1: router (logits -> softmax -> top-2 -> renorm) ----------

__global__ __launch_bounds__(256)
void moe_router_kernel(const float* __restrict__ x, const float* __restrict__ wg,
                       int* __restrict__ topk_idx, float* __restrict__ topk_w) {
  const int token = blockIdx.x * 8 + (threadIdx.x >> 5);
  const int lane  = threadIdx.x & 31;
  float acc[kE];
#pragma unroll
  for (int e = 0; e < kE; ++e) acc[e] = 0.0f;
  for (int h = lane; h < kH; h += 32) {
    const float xv = x[(size_t)token * kH + h];
#pragma unroll
    for (int e = 0; e < kE; ++e) acc[e] += xv * wg[(size_t)e * kH + h];
  }
#pragma unroll
  for (int e = 0; e < kE; ++e) {
#pragma unroll
    for (int off = 16; off > 0; off >>= 1)
      acc[e] += __shfl_xor(acc[e], off, 32);
  }
  if (lane == 0) {
    float mx = acc[0];
#pragma unroll
    for (int e = 1; e < kE; ++e) mx = fmaxf(mx, acc[e]);
    float p[kE], s = 0.0f;
#pragma unroll
    for (int e = 0; e < kE; ++e) { p[e] = __expf(acc[e] - mx); s += p[e]; }
    const float inv = 1.0f / s;
#pragma unroll
    for (int e = 0; e < kE; ++e) p[e] *= inv;
    int i0 = 0;
#pragma unroll
    for (int e = 1; e < kE; ++e) if (p[e] > p[i0]) i0 = e;   // lowest index on tie
    int i1 = (i0 == 0) ? 1 : 0;
#pragma unroll
    for (int e = 0; e < kE; ++e) if (e != i0 && p[e] > p[i1]) i1 = e;
    const float wsum = p[i0] + p[i1];
    topk_idx[2 * token]     = i0;
    topk_idx[2 * token + 1] = i1;
    topk_w[2 * token]       = p[i0] / wsum;
    topk_w[2 * token + 1]   = p[i1] / wsum;
  }
}

// ---------- kernel 2/3: zero counts, scatter token rows per expert ----------

__global__ void moe_zero_kernel(int* __restrict__ counts) {
  if (threadIdx.x < kE) counts[threadIdx.x] = 0;
}

__global__ __launch_bounds__(256)
void moe_scatter_kernel(const int* __restrict__ topk_idx,
                        int* __restrict__ counts, int* __restrict__ rows) {
  const int t = blockIdx.x * 256 + threadIdx.x;
  if (t >= kT) return;
#pragma unroll
  for (int k = 0; k < 2; ++k) {
    const int e = topk_idx[2 * t + k];
    const int pos = atomicAdd(&counts[e], 1);
    rows[(size_t)e * kT + pos] = 2 * t + k;    // destination row id (token,slot)
  }
}

// ---------- kernel 4: fused expert (GEMM1 + SwiGLU + GEMM2), all-bf16 loads ----------

__global__ __launch_bounds__(256)
void moe_expert_kernel(const unsigned short* __restrict__ xbf,
                       const unsigned short* __restrict__ w1bf,
                       const unsigned short* __restrict__ w3bf,
                       const unsigned short* __restrict__ w2bf,
                       const int* __restrict__ counts,
                       const int* __restrict__ rows,
                       float* __restrict__ yc) {
  extern __shared__ unsigned short smem[];
  unsigned short* xt  = smem;                        // [kM][kLdx] bf16 : 64.5 KB
  unsigned short* act = smem + kM * kLdx;            // [kM][kLda] bf16 : 128.5 KB
  int* rowdest = (int*)(smem + kM * kLdx + kM * kLda); // [kM] dest row ids

  const int e    = blockIdx.x >> 7;     // expert
  const int tile = blockIdx.x & 127;    // token tile within expert
  const int ne   = counts[e];
  const int base = tile * kM;
  if (base >= ne) return;               // block-uniform early exit

  const int tid  = threadIdx.x;
  const int wave = tid >> 5;
  const int lane = tid & 31;
  const int rlo  = lane & 15;           // WMMA row (A) / col (B,C) index
  const int mofs = (lane < 16) ? 0 : 8; // C-fragment row offset

  // phase 0: destination rows for this tile (pad rows duplicate row 0)
  if (tid < kM) {
    const int idx = base + tid;
    rowdest[tid] = rows[(size_t)e * kT + ((idx < ne) ? idx : base)];
  }
  __syncthreads();

  // phase 1: gather bf16 X tile -> LDS (pure b128 copy, no conversion)
  // tile = 32 rows * 128 uint4; 256 threads -> 16 each
  for (int it = 0; it < 16; ++it) {
    const int q = it * 256 + tid;       // uint4 index inside tile
    const int r = q >> 7;               // 128 uint4 per row
    const int c = q & 127;
    const int tok = rowdest[r] >> 1;    // token id
    *(uint4*)(xt + r * kLdx + c * 8) =
        *(const uint4*)(xbf + (size_t)tok * kH + c * 8);
  }
  __syncthreads();

  // phase 2: h1 = X@w1^T, h3 = X@w3^T, act = silu(h1)*h3  -> LDS bf16
  const unsigned short* w1e = w1bf + (size_t)e * kI * kH;
  const unsigned short* w3e = w3bf + (size_t)e * kI * kH;
  for (int nt = wave; nt < (kI / 16); nt += 8) {
    const int n0 = nt * 16;
    v8f c1a = {}, c1b = {}, c3a = {}, c3b = {};
    const unsigned short* b1row = w1e + (size_t)(n0 + rlo) * kH;
    const unsigned short* b3row = w3e + (size_t)(n0 + rlo) * kH;
    if (nt + 8 < (kI / 16)) {           // pull next tile's weight rows toward WGP
      __builtin_prefetch(b1row + (size_t)128 * kH, 0, 3);
      __builtin_prefetch(b3row + (size_t)128 * kH, 0, 3);
    }
#pragma unroll 2
    for (int kb = 0; kb < kH; kb += 32) {
      const v8u a0  = load_a_lds(xt, kLdx, rlo,      kb, lane);
      const v8u a1  = load_a_lds(xt, kLdx, rlo + 16, kb, lane);
      const v8u b1f = load_b_bf(b1row, kb, lane);
      const v8u b3f = load_b_bf(b3row, kb, lane);
      c1a = wmma_bf16(a0, b1f, c1a);
      c1b = wmma_bf16(a1, b1f, c1b);
      c3a = wmma_bf16(a0, b3f, c3a);
      c3b = wmma_bf16(a1, b3f, c3b);
    }
    const int col = n0 + rlo;
#pragma unroll
    for (int v = 0; v < 8; ++v) {
      const float g0 = c1a[v], u0 = c3a[v];
      const float g1 = c1b[v], u1 = c3b[v];
      const float s0 = g0 * fast_rcp(1.0f + __expf(-g0)) * u0;   // silu(g)*u
      const float s1 = g1 * fast_rcp(1.0f + __expf(-g1)) * u1;
      act[(size_t)(v + mofs) * kLda + col]      = f2bf(s0);
      act[(size_t)(16 + v + mofs) * kLda + col] = f2bf(s1);
    }
  }
  __syncthreads();

  // phase 3: Y = act @ w2^T  -> ycontrib rows (write-once, deterministic)
  const unsigned short* w2e = w2bf + (size_t)e * kH * kI;
  for (int nt = wave; nt < (kH / 16); nt += 8) {
    const int n0 = nt * 16;
    v8f ca = {}, cb = {};
    const unsigned short* b2row = w2e + (size_t)(n0 + rlo) * kI;
    if (nt + 8 < (kH / 16)) {
      __builtin_prefetch(b2row + (size_t)128 * kI, 0, 3);
    }
#pragma unroll 2
    for (int kb = 0; kb < kI; kb += 32) {
      const v8u a0  = load_a_lds(act, kLda, rlo,      kb, lane);
      const v8u a1  = load_a_lds(act, kLda, rlo + 16, kb, lane);
      const v8u b2f = load_b_bf(b2row, kb, lane);
      ca = wmma_bf16(a0, b2f, ca);
      cb = wmma_bf16(a1, b2f, cb);
    }
    const int col = n0 + rlo;
#pragma unroll
    for (int v = 0; v < 8; ++v) {
      const int m0 = v + mofs;
      const int m1 = 16 + v + mofs;
      if (base + m0 < ne) yc[(size_t)rowdest[m0] * kH + col] = ca[v];
      if (base + m1 < ne) yc[(size_t)rowdest[m1] * kH + col] = cb[v];
    }
  }
}

// ---------- kernel 5: weighted combine of the two slot contributions ----------

__global__ __launch_bounds__(256)
void moe_combine_kernel(const float* __restrict__ yc,
                        const float* __restrict__ topk_w,
                        float* __restrict__ out) {
  const size_t i = (size_t)blockIdx.x * 256 + threadIdx.x;  // < kT*kH
  const size_t t = i >> 10;          // / kH
  const size_t h = i & (kH - 1);
  out[i] = topk_w[2 * t]     * yc[(2 * t)     * kH + h] +
           topk_w[2 * t + 1] * yc[(2 * t + 1) * kH + h];
}

// ---------- launch ----------

extern "C" void kernel_launch(void* const* d_in, const int* in_sizes, int n_in,
                              void* d_out, int out_size, void* d_ws, size_t ws_size,
                              hipStream_t stream) {
  (void)in_sizes; (void)n_in; (void)out_size; (void)ws_size;
  const float* x  = (const float*)d_in[0];  // [1,4096,1024]
  const float* wg = (const float*)d_in[1];  // [8,1024]
  const float* w1 = (const float*)d_in[2];  // [8,2048,1024]
  const float* w3 = (const float*)d_in[3];  // [8,2048,1024]
  const float* w2 = (const float*)d_in[4];  // [8,1024,2048]
  float* out = (float*)d_out;               // [1,4096,1024]

  char* ws = (char*)d_ws;
  size_t o = 0;
  int*   topk_idx = (int*)(ws + o);   o += (size_t)kT * 2 * sizeof(int);        // 32 KB
  float* topk_w   = (float*)(ws + o); o += (size_t)kT * 2 * sizeof(float);      // 32 KB
  int*   counts   = (int*)(ws + o);   o += 256;                                 // padded
  int*   rows     = (int*)(ws + o);   o += (size_t)kE * kT * sizeof(int);       // 128 KB
  float* yc       = (float*)(ws + o); o += (size_t)kT * 2 * kH * sizeof(float); // 32 MB
  unsigned short* xbf  = (unsigned short*)(ws + o); o += (size_t)kT * kH * 2;       // 8 MB
  unsigned short* w1bf = (unsigned short*)(ws + o); o += (size_t)kE * kI * kH * 2;  // 32 MB
  unsigned short* w3bf = (unsigned short*)(ws + o); o += (size_t)kE * kI * kH * 2;  // 32 MB
  unsigned short* w2bf = (unsigned short*)(ws + o); o += (size_t)kE * kH * kI * 2;  // 32 MB

  // one-pass fp32 -> bf16 conversions (hoists all conversion VALU out of GEMMs)
  const int nX4 = (kT * kH) / 4;
  const int nW4 = (kE * kI * kH) / 4;
  moe_cvt_kernel<<<(nX4 + 255) / 256, 256, 0, stream>>>(x,  xbf,  nX4);
  moe_cvt_kernel<<<(nW4 + 255) / 256, 256, 0, stream>>>(w1, w1bf, nW4);
  moe_cvt_kernel<<<(nW4 + 255) / 256, 256, 0, stream>>>(w3, w3bf, nW4);
  moe_cvt_kernel<<<(nW4 + 255) / 256, 256, 0, stream>>>(w2, w2bf, nW4);

  moe_zero_kernel<<<1, 64, 0, stream>>>(counts);
  moe_router_kernel<<<kT / 8, 256, 0, stream>>>(x, wg, topk_idx, topk_w);
  moe_scatter_kernel<<<kT / 256, 256, 0, stream>>>(topk_idx, counts, rows);

  const size_t smem = (size_t)(kM * kLdx + kM * kLda) * sizeof(unsigned short) +
                      (size_t)kM * sizeof(int);   // ~197.8 KB (<= 320 KB WGP LDS)
  moe_expert_kernel<<<kE * (kT / kM), 256, smem, stream>>>(xbf, w1bf, w3bf, w2bf,
                                                           counts, rows, yc);

  moe_combine_kernel<<<(kT * kH) / 256, 256, 0, stream>>>(yc, topk_w, out);
}
